// Crossformer_38319698214993
// MI455X (gfx1250) — matrix operand order repaired and checked by reading, not compiled, not run on hardware
//
#include <hip/hip_runtime.h>
#include <hip/hip_bf16.h>
#include <cstdint>
#include <cstddef>

// ---------------------------------------------------------------------------
// CDNA5 (gfx1250) wave32 WMMA types & fragment helpers
// ---------------------------------------------------------------------------
typedef __attribute__((ext_vector_type(16))) __bf16 bvec16;
typedef __attribute__((ext_vector_type(8)))  float  fvec8;
typedef int vi4 __attribute__((vector_size(16)));   // int4 for async-LDS builtin

__device__ __forceinline__ fvec8 wmma_bf16(bvec16 a, bvec16 b, fvec8 c) {
  // D(16x16,f32) = A(16x32,bf16) * B(32x16,bf16) + C
  return __builtin_amdgcn_wmma_f32_16x16x32_bf16(false, a, false, b, (short)0, c,
                                                 false, false);
}

// A-fragment (16x32 bf16). 'row' points at A[m][k0] for this lane's row m.
__device__ __forceinline__ bvec16 frag_a(const __bf16* row, int hi) {
  union { bvec16 v; uint32_t u[8]; } r;
#pragma unroll
  for (int i = 0; i < 8; ++i) {
    const int k = ((i & 3) * 2) + ((i >> 2) * 16) + hi * 8;
    r.u[i] = *(const uint32_t*)(row + k);
  }
  return r.v;
}

// B-fragment (32x16 bf16) gathered from K-contiguous storage:
// 'rowN' points at Bt[n][k0] where Bt[n][k] == B[k][n].
__device__ __forceinline__ bvec16 frag_b(const __bf16* rowN, int hi) {
  union { bvec16 v; uint32_t u[8]; } r;
#pragma unroll
  for (int i = 0; i < 8; ++i) {
    const int k = ((i & 3) * 2) + ((i >> 2) * 16) + hi * 8;
    r.u[i] = *(const uint32_t*)(rowN + k);
  }
  return r.v;
}

// ---------------------------------------------------------------------------
// gfx1250 async global->LDS copy (16B per lane), with sync fallback.
// Param types per hipcc diagnostic: (AS1 int4*, AS3 int4*, imm, imm).
// ---------------------------------------------------------------------------
__device__ __forceinline__ void async_copy_b128(const __bf16* g, __bf16* l) {
#if defined(__HIP_DEVICE_COMPILE__) && \
    __has_builtin(__builtin_amdgcn_global_load_async_to_lds_b128)
  __builtin_amdgcn_global_load_async_to_lds_b128(
      (__attribute__((address_space(1))) vi4*)g,
      (__attribute__((address_space(3))) vi4*)l, 0, 0);
#else
  *(uint4*)l = *(const uint4*)g;
#endif
}

__device__ __forceinline__ void async_wait_all() {
#if defined(__HIP_DEVICE_COMPILE__) && \
    __has_builtin(__builtin_amdgcn_global_load_async_to_lds_b128)
#if __has_builtin(__builtin_amdgcn_s_wait_asynccnt)
  __builtin_amdgcn_s_wait_asynccnt(0);
#else
  asm volatile("s_wait_asynccnt 0" ::: "memory");
#endif
#endif
}

__device__ __forceinline__ float gelu_tanh(float x) {
  const float c0 = 0.7978845608028654f;   // sqrt(2/pi)
  const float c1 = 0.044715f;
  float t = tanhf(c0 * (x + c1 * x * x * x));
  return 0.5f * x * (1.0f + t);
}

// ---------------------------------------------------------------------------
// LayerNorm (f32 in) -> bf16 out.  One block per row, 256 threads (8 waves).
// ---------------------------------------------------------------------------
__global__ __launch_bounds__(256)
void layernorm_bf16_kernel(const float* __restrict__ X, const float* __restrict__ w,
                           const float* __restrict__ b, __bf16* __restrict__ Y,
                           int cols) {
  __shared__ float sred[8];
  __shared__ float smean, srstd;
  const int row = blockIdx.x;
  const float* x = X + (size_t)row * cols;
  const int lane = threadIdx.x & 31, wid = threadIdx.x >> 5;

  float s = 0.f;
  for (int i = threadIdx.x; i < cols; i += 256) s += x[i];
#pragma unroll
  for (int m = 16; m; m >>= 1) s += __shfl_xor(s, m, 32);
  if (lane == 0) sred[wid] = s;
  __syncthreads();
  if (threadIdx.x == 0) {
    float t = 0.f;
#pragma unroll
    for (int i = 0; i < 8; ++i) t += sred[i];
    smean = t / (float)cols;
  }
  __syncthreads();
  const float mean = smean;

  float v = 0.f;
  for (int i = threadIdx.x; i < cols; i += 256) {
    float c = x[i] - mean;
    v += c * c;
  }
#pragma unroll
  for (int m = 16; m; m >>= 1) v += __shfl_xor(v, m, 32);
  if (lane == 0) sred[wid] = v;
  __syncthreads();
  if (threadIdx.x == 0) {
    float t = 0.f;
#pragma unroll
    for (int i = 0; i < 8; ++i) t += sred[i];
    srstd = rsqrtf(t / (float)cols + 1e-12f);
  }
  __syncthreads();
  const float rstd = srstd;

  __bf16* y = Y + (size_t)row * cols;
  for (int i = threadIdx.x; i < cols; i += 256)
    y[i] = (__bf16)((x[i] - mean) * rstd * w[i] + b[i]);
}

// ---------------------------------------------------------------------------
// f32 -> bf16 elementwise conversion
// ---------------------------------------------------------------------------
__global__ __launch_bounds__(256)
void cvt_bf16_kernel(const float* __restrict__ src, __bf16* __restrict__ dst, int n) {
  int i = blockIdx.x * 256 + threadIdx.x;
  if (i < n) dst[i] = (__bf16)src[i];
}

// ---------------------------------------------------------------------------
// bf16 WMMA GEMM: C = epilogue(A[M,K] * B[K,N] + bias [+ resid])
// 128x128 block tile, 256 threads (8 waves); each wave owns a 32x64 output
// (2 A-frags x 4 B-frags -> 8 WMMA per 32-deep K-step).  Double-buffered LDS,
// async A-tile staging, single barrier per K-step.
// Requires M%128==0, N%128==0, K%32==0.
// ---------------------------------------------------------------------------
template <bool GELU, bool RESID, typename OutT>
__global__ __launch_bounds__(256)
void gemm_bf16_kernel(const __bf16* __restrict__ A, const __bf16* __restrict__ B,
                      const float* __restrict__ bias, const float* __restrict__ resid,
                      OutT* __restrict__ C, int Mdim, int Ndim, int Kdim) {
  __shared__ alignas(16) __bf16 ldsA[2][128 * 32];   // [m][k]
  __shared__ alignas(16) __bf16 ldsBt[2][128 * 32];  // [n][k]  (B transposed)

  const int tid = threadIdx.x;
  const int lane = tid & 31, wid = tid >> 5;
  const int hl = lane & 15, hi = lane >> 4;
  const int bm = blockIdx.x * 128;
  const int bn = blockIdx.y * 128;
  const int rowg = (wid & 3) * 32;       // wave's 32-row group
  const int colg = (wid >> 2) * 64;      // wave's 64-col group

  // per-thread staging coordinates
  const int ar = tid >> 1, ac = (tid & 1) * 16;          // A: 128x32, 2x16B
  const int bkr = tid >> 3, bc = (tid & 7) * 16;         // B: 32x128, 2x16B
  const __bf16* Ap = A + (size_t)(bm + ar) * Kdim + ac;
  const __bf16* Bp = B + (size_t)bkr * Ndim + bn + bc;

  fvec8 acc[2][4];
#pragma unroll
  for (int i = 0; i < 2; ++i)
#pragma unroll
    for (int j = 0; j < 4; ++j) acc[i][j] = fvec8{};

  // ---- prologue: stage K-tile 0 into buffer 0 ----
  {
    async_copy_b128(Ap, &ldsA[0][ar * 32 + ac]);
    async_copy_b128(Ap + 8, &ldsA[0][ar * 32 + ac + 8]);
    union { uint4 u[2]; __bf16 h[16]; } rb;
    rb.u[0] = *(const uint4*)Bp;
    rb.u[1] = *(const uint4*)(Bp + 8);
#pragma unroll
    for (int e = 0; e < 16; ++e) ldsBt[0][(bc + e) * 32 + bkr] = rb.h[e];
  }
  async_wait_all();
  __syncthreads();

  int cur = 0;
  for (int k0 = 0; k0 < Kdim; k0 += 32) {
    const int nxt = cur ^ 1;
    const bool has_next = (k0 + 32) < Kdim;

    union { uint4 u[2]; __bf16 h[16]; } rb;
    if (has_next) {  // issue next-tile loads (A async to LDS, B to regs)
      const __bf16* Apn = Ap + (k0 + 32);
      async_copy_b128(Apn, &ldsA[nxt][ar * 32 + ac]);
      async_copy_b128(Apn + 8, &ldsA[nxt][ar * 32 + ac + 8]);
      const __bf16* Bpn = Bp + (size_t)(k0 + 32) * Ndim;
      rb.u[0] = *(const uint4*)Bpn;
      rb.u[1] = *(const uint4*)(Bpn + 8);
    }

    // ---- 8 WMMAs on the current tile ----
    bvec16 fa0 = frag_a(&ldsA[cur][(rowg + hl) * 32], hi);
    bvec16 fa1 = frag_a(&ldsA[cur][(rowg + 16 + hl) * 32], hi);
#pragma unroll
    for (int j = 0; j < 4; ++j) {
      bvec16 fb = frag_b(&ldsBt[cur][(colg + j * 16 + hl) * 32], hi);
      acc[0][j] = wmma_bf16(fa0, fb, acc[0][j]);
      acc[1][j] = wmma_bf16(fa1, fb, acc[1][j]);
    }

    if (has_next) {  // scatter-transpose B into the other buffer
#pragma unroll
      for (int e = 0; e < 16; ++e) ldsBt[nxt][(bc + e) * 32 + bkr] = rb.h[e];
    }
    async_wait_all();
    __syncthreads();
    cur = nxt;
  }

  // ---- epilogue ----
#pragma unroll
  for (int i = 0; i < 2; ++i) {
    const int growBase = bm + rowg + i * 16 + hi * 8;
#pragma unroll
    for (int j = 0; j < 4; ++j) {
      const int n = bn + colg + j * 16 + hl;
      const float bval = bias[n];
#pragma unroll
      for (int r = 0; r < 8; ++r) {
        const int m = growBase + r;
        float v = acc[i][j][r] + bval;
        if constexpr (GELU) v = gelu_tanh(v);
        if constexpr (RESID) v += resid[(size_t)m * Ndim + n];
        C[(size_t)m * Ndim + n] = (OutT)v;
      }
    }
  }
}

// ---------------------------------------------------------------------------
// Flash attention, HD=64, bf16 in/out, f32 accumulate.
// grid = (Nq/64, H), block = 128 threads (4 waves); each wave = 16 query rows.
// Q,K,V,O are [tokens, 1024] with head h at column offset h*64.
// ---------------------------------------------------------------------------
__global__ __launch_bounds__(128)
void flash_attn_kernel(const __bf16* __restrict__ Q, const __bf16* __restrict__ K,
                       const __bf16* __restrict__ V, __bf16* __restrict__ O,
                       int Mctx) {
  constexpr int Dm = 1024;
  __shared__ alignas(16) __bf16 ldsK[64 * 64];    // [key][d]   == Bt for Q*K^T
  __shared__ alignas(16) __bf16 ldsVt[64 * 64];   // [d][key]   == Bt for P*V
  __shared__ alignas(16) __bf16 ldsP[4][16 * 64]; // per-wave P staging

  const int tid = threadIdx.x;
  const int lane = tid & 31, wid = tid >> 5;
  const int hl = lane & 15, hi = lane >> 4;
  const int hoff = blockIdx.y * 64;
  const int qbase = blockIdx.x * 64 + wid * 16;

  const __bf16* qrow = Q + (size_t)(qbase + hl) * Dm + hoff;
  const bvec16 fq0 = frag_a(qrow, hi);       // K-dim 0..31
  const bvec16 fq1 = frag_a(qrow + 32, hi);  // K-dim 32..63

  float mrow[8], lrow[8];
  fvec8 o[4] = {fvec8{}, fvec8{}, fvec8{}, fvec8{}};
#pragma unroll
  for (int r = 0; r < 8; ++r) { mrow[r] = -1e30f; lrow[r] = 0.f; }

  const int key = tid >> 1, d0 = (tid & 1) * 32;

  for (int j0 = 0; j0 < Mctx; j0 += 64) {
    // ---- stage 64-key chunk: K async to LDS (used row-major), V transposed ----
    {
      const __bf16* ksrc = K + (size_t)(j0 + key) * Dm + hoff + d0;
      __bf16* kdst = &ldsK[key * 64 + d0];
#pragma unroll
      for (int i = 0; i < 4; ++i) async_copy_b128(ksrc + i * 8, kdst + i * 8);

      const __bf16* vsrc = V + (size_t)(j0 + key) * Dm + hoff + d0;
#pragma unroll
      for (int i = 0; i < 4; ++i) {
        union { uint4 u; __bf16 h[8]; } d;
        d.u = *(const uint4*)(vsrc + i * 8);
#pragma unroll
        for (int q = 0; q < 8; ++q) ldsVt[(d0 + i * 8 + q) * 64 + key] = d.h[q];
      }
      if (j0 + 64 < Mctx) {  // prefetch next chunk (global_prefetch_b8)
        __builtin_prefetch(V + (size_t)(j0 + 64 + key) * Dm + hoff + d0, 0, 0);
        __builtin_prefetch(K + (size_t)(j0 + 64 + key) * Dm + hoff + d0, 0, 0);
      }
    }
    async_wait_all();
    __syncthreads();

    // ---- S = (Q * K^T) * scale : four 16x16 tiles along the 64 keys ----
    fvec8 s[4];
#pragma unroll
    for (int c = 0; c < 4; ++c) {
      const __bf16* krow = &ldsK[(c * 16 + hl) * 64];
      fvec8 a = {};
      a = wmma_bf16(fq0, frag_b(krow, hi), a);
      a = wmma_bf16(fq1, frag_b(krow + 32, hi), a);
      s[c] = a;
    }

    // ---- online softmax (row stats across 16-lane half-groups) ----
#pragma unroll
    for (int r = 0; r < 8; ++r) {
      float lm = -1e30f;
#pragma unroll
      for (int c = 0; c < 4; ++c) {
        s[c][r] *= 0.125f;  // 1/sqrt(64)
        lm = fmaxf(lm, s[c][r]);
      }
#pragma unroll
      for (int msk = 8; msk; msk >>= 1) lm = fmaxf(lm, __shfl_xor(lm, msk, 32));
      const float mnew = fmaxf(mrow[r], lm);
      const float alpha = __expf(mrow[r] - mnew);
      mrow[r] = mnew;
      float rs = 0.f;
#pragma unroll
      for (int c = 0; c < 4; ++c) {
        float p = __expf(s[c][r] - mnew);
        s[c][r] = p;
        rs += p;
      }
#pragma unroll
      for (int msk = 8; msk; msk >>= 1) rs += __shfl_xor(rs, msk, 32);
      lrow[r] = lrow[r] * alpha + rs;
#pragma unroll
      for (int c = 0; c < 4; ++c) o[c][r] *= alpha;
    }

    // ---- C-layout -> A-layout via per-wave LDS, then O += P * V ----
    __bf16* pw = &ldsP[wid][0];
#pragma unroll
    for (int c = 0; c < 4; ++c)
#pragma unroll
      for (int r = 0; r < 8; ++r)
        pw[(hi * 8 + r) * 64 + c * 16 + hl] = (__bf16)s[c][r];
    __builtin_amdgcn_wave_barrier();  // keep DS stores before the reloads

    const bvec16 fp0 = frag_a(&pw[hl * 64], hi);
    const bvec16 fp1 = frag_a(&pw[hl * 64] + 32, hi);
#pragma unroll
    for (int c = 0; c < 4; ++c) {
      const __bf16* vrow = &ldsVt[(c * 16 + hl) * 64];
      o[c] = wmma_bf16(fp0, frag_b(vrow, hi), o[c]);
      o[c] = wmma_bf16(fp1, frag_b(vrow + 32, hi), o[c]);
    }
    __syncthreads();  // before next chunk overwrites ldsK/ldsVt
  }

  // ---- normalize and store bf16 ----
#pragma unroll
  for (int c = 0; c < 4; ++c)
#pragma unroll
    for (int r = 0; r < 8; ++r) {
      const int m = qbase + hi * 8 + r;
      const int d = hoff + c * 16 + hl;
      O[(size_t)m * Dm + d] = (__bf16)(o[c][r] / lrow[r]);
    }
}

// ---------------------------------------------------------------------------
// Host launcher
// ---------------------------------------------------------------------------
extern "C" void kernel_launch(void* const* d_in, const int* in_sizes, int n_in,
                              void* d_out, int out_size, void* d_ws, size_t ws_size,
                              hipStream_t stream) {
  (void)in_sizes; (void)n_in; (void)out_size; (void)ws_size;
  constexpr int N = 2048, M = 2048, D = 1024, C = 768, H = 16, D4 = 4096;

  const float* x     = (const float*)d_in[0];
  const float* ctx   = (const float*)d_in[1];
  const float* wq    = (const float*)d_in[2];
  const float* bq    = (const float*)d_in[3];
  const float* wk    = (const float*)d_in[4];
  const float* bk    = (const float*)d_in[5];
  const float* wv    = (const float*)d_in[6];
  const float* bv    = (const float*)d_in[7];
  const float* wo    = (const float*)d_in[8];
  const float* bo    = (const float*)d_in[9];
  const float* w1    = (const float*)d_in[10];
  const float* b1    = (const float*)d_in[11];
  const float* w2    = (const float*)d_in[12];
  const float* b2    = (const float*)d_in[13];
  const float* qn_w  = (const float*)d_in[14];
  const float* qn_b  = (const float*)d_in[15];
  const float* kvn_w = (const float*)d_in[16];
  const float* kvn_b = (const float*)d_in[17];
  const float* pn_w  = (const float*)d_in[18];
  const float* pn_b  = (const float*)d_in[19];
  float* out = (float*)d_out;

  char* p = (char*)d_ws;
  auto alloc = [&](size_t bytes) -> char* {
    char* r = p;
    p += (bytes + 255) & ~(size_t)255;
    return r;
  };
  __bf16* xq_bf = (__bf16*)alloc((size_t)N * D * 2);
  __bf16* kv_bf = (__bf16*)alloc((size_t)M * C * 2);
  __bf16* wq_bf = (__bf16*)alloc((size_t)D * D * 2);
  __bf16* wk_bf = (__bf16*)alloc((size_t)C * D * 2);
  __bf16* wv_bf = (__bf16*)alloc((size_t)C * D * 2);
  __bf16* wo_bf = (__bf16*)alloc((size_t)D * D * 2);
  __bf16* w1_bf = (__bf16*)alloc((size_t)D * D4 * 2);
  __bf16* w2_bf = (__bf16*)alloc((size_t)D4 * D * 2);
  __bf16* q_bf  = (__bf16*)alloc((size_t)N * D * 2);
  __bf16* k_bf  = (__bf16*)alloc((size_t)M * D * 2);
  __bf16* v_bf  = (__bf16*)alloc((size_t)M * D * 2);
  __bf16* o_bf  = (__bf16*)alloc((size_t)N * D * 2);
  float*  x2    = (float*) alloc((size_t)N * D * 4);
  __bf16* h1_bf = (__bf16*)alloc((size_t)N * D * 2);
  __bf16* h2_bf = (__bf16*)alloc((size_t)N * D4 * 2);

  // 1) pre-norms -> bf16
  layernorm_bf16_kernel<<<dim3(N), dim3(256), 0, stream>>>(x, qn_w, qn_b, xq_bf, D);
  layernorm_bf16_kernel<<<dim3(M), dim3(256), 0, stream>>>(ctx, kvn_w, kvn_b, kv_bf, C);

  // 2) weights -> bf16
  auto cvt = [&](const float* s, __bf16* d, int n) {
    cvt_bf16_kernel<<<dim3((n + 255) / 256), dim3(256), 0, stream>>>(s, d, n);
  };
  cvt(wq, wq_bf, D * D);
  cvt(wk, wk_bf, C * D);
  cvt(wv, wv_bf, C * D);
  cvt(wo, wo_bf, D * D);
  cvt(w1, w1_bf, D * D4);
  cvt(w2, w2_bf, D4 * D);

  // 3) Q/K/V projections (WMMA)
  gemm_bf16_kernel<false, false, __bf16>
      <<<dim3(N / 128, D / 128), dim3(256), 0, stream>>>(xq_bf, wq_bf, bq, nullptr,
                                                         q_bf, N, D, D);
  gemm_bf16_kernel<false, false, __bf16>
      <<<dim3(M / 128, D / 128), dim3(256), 0, stream>>>(kv_bf, wk_bf, bk, nullptr,
                                                         k_bf, M, D, C);
  gemm_bf16_kernel<false, false, __bf16>
      <<<dim3(M / 128, D / 128), dim3(256), 0, stream>>>(kv_bf, wv_bf, bv, nullptr,
                                                         v_bf, M, D, C);

  // 4) flash attention (WMMA + online softmax)
  flash_attn_kernel<<<dim3(N / 64, H), dim3(128), 0, stream>>>(q_bf, k_bf, v_bf,
                                                               o_bf, M);

  // 5) output projection + residual (f32 out)
  gemm_bf16_kernel<false, true, float>
      <<<dim3(N / 128, D / 128), dim3(256), 0, stream>>>(o_bf, wo_bf, bo, x, x2,
                                                         N, D, D);

  // 6) MLP: ln -> (W1,gelu) -> (W2,+resid) -> d_out
  layernorm_bf16_kernel<<<dim3(N), dim3(256), 0, stream>>>(x2, pn_w, pn_b, h1_bf, D);
  gemm_bf16_kernel<true, false, __bf16>
      <<<dim3(N / 128, D4 / 128), dim3(256), 0, stream>>>(h1_bf, w1_bf, b1, nullptr,
                                                          h2_bf, N, D4, D);
  gemm_bf16_kernel<false, true, float>
      <<<dim3(N / 128, D / 128), dim3(256), 0, stream>>>(h2_bf, w2_bf, b2, x2, out,
                                                         N, D, D4);
}
